// MAB_58480274702998
// MI455X (gfx1250) — matrix operand order repaired and checked by reading, not compiled, not run on hardware
//
#include <hip/hip_runtime.h>

// ---------------------------------------------------------------------------
// MAB (Set-Transformer multihead attention block) for gfx1250 / MI455X.
// All GEMMs + attention via v_wmma_f32_16x16x32_f16 (wave32 WMMA).
// LDS-staged operands (double-buffered K/V tiles in attention, pre-converted
// A panel in the GEMMs); softmax row-sums accumulated on the matrix pipe;
// raw v_exp_f32 for the softmax exponentials.
// B=4, NQ=NK=2048, DIM=512, heads=8, dh=64.
// ---------------------------------------------------------------------------

typedef _Float16 v16h __attribute__((ext_vector_type(16)));
typedef _Float16 h8   __attribute__((ext_vector_type(8)));
typedef float    v8f  __attribute__((ext_vector_type(8)));
typedef float    f4   __attribute__((ext_vector_type(4)));

#define SEQ    2048
#define DMODEL 512
#define NTOK   8192   // B * SEQ
#define NHEAD  8
#define DHEAD  64

// LDS strides (halves) padded for bank-conflict-free 16-row-strided reads.
#define KSTR 72   // K tile rows: 64 halves + 8 pad
#define VSTR 40   // V tile rows: 32 halves + 8 pad
#define ASTR 520  // A panel rows: 512 halves + 8 pad

// log2(e) / sqrt(512): folded into the q-projection f16 store so the
// attention kernel can use raw exp2 and skip all per-score scaling.
#define QSCALE (1.4426950408889634f / 22.627416997969522f)

__device__ __forceinline__ v8f wmma_f16(v16h a, v16h b, v8f c) {
  // (neg_a, A, neg_b, B, c_mod, C, reuse_a, reuse_b)
  return __builtin_amdgcn_wmma_f32_16x16x32_f16(false, a, false, b, (short)0, c,
                                                false, false);
}

__device__ __forceinline__ v16h h8cat(h8 lo, h8 hi) {
  v16h r;
#pragma unroll
  for (int i = 0; i < 8; ++i) { r[i] = lo[i]; r[i + 8] = hi[i]; }
  return r;
}

__device__ __forceinline__ v8f vzero8() {
  v8f z;
#pragma unroll
  for (int i = 0; i < 8; ++i) z[i] = 0.0f;
  return z;
}

__device__ __forceinline__ bool any_lane(bool p) {
#if __has_builtin(__builtin_amdgcn_ballot_w32)
  return __builtin_amdgcn_ballot_w32(p) != 0u;
#else
  return __ballot(p) != 0ull;
#endif
}

// Raw v_exp_f32: no denormal-range fixup (softmax args are <= 0; inputs below
// -126 correctly flush to 0). Single TRANS op, co-executes with WMMA.
__device__ __forceinline__ float fast_exp2(float x) {
#if __has_builtin(__builtin_amdgcn_exp2f)
  return __builtin_amdgcn_exp2f(x);
#else
  return exp2f(x);
#endif
}

// ---------------------------------------------------------------------------
// Weight convert: W [k,n] f32 row-major -> Wt [n,k] f16 (n-major). 512x512.
// ---------------------------------------------------------------------------
__global__ __launch_bounds__(256) void wconv_kernel(const float* __restrict__ W,
                                                    _Float16* __restrict__ Wt) {
  int idx = blockIdx.x * 256 + threadIdx.x;   // 0 .. 512*512-1
  int n = idx >> 9, k = idx & 511;
  Wt[(size_t)n * 512 + k] = (_Float16)W[(size_t)k * 512 + n];
}

// ---------------------------------------------------------------------------
// GEMM: out[m,n] = X[m,:] @ Wt[n,:] + bias[n], M=8192, N=K=512.
// One block = 32 rows staged once into LDS as f16 (conversion done once, no
// redundant per-wave L2 reads); 8 waves each own a 32(M)x64(N) tile.
// MODE 0: q proj   -> outH f16 row-major (scaled by QSCALE) + outF f32 copy
// MODE 1: k proj   -> outH f16 row-major
// MODE 2: v proj   -> outH f16 TRANSPOSED per batch: vT[b][n][token]
// MODE 3: FFN      -> outF = resF + relu(acc + bias), A read from f16 Xh
// ---------------------------------------------------------------------------
template <int MODE>
__global__ __launch_bounds__(256) void gemm512_kernel(
    const float* __restrict__ Xf, const _Float16* __restrict__ Xh,
    const _Float16* __restrict__ Wt, const float* __restrict__ bias,
    _Float16* __restrict__ outH, float* __restrict__ outF,
    const float* __restrict__ resF) {
  __shared__ _Float16 sA[32 * ASTR];
  const int tid  = threadIdx.x;
  const int lane = tid & 31;
  const int wave = tid >> 5;
  const int g  = lane >> 4;       // lane half (0/1)
  const int ln = lane & 15;
  const int m0 = blockIdx.x * 32; // row tile (two 16-row blocks)
  const int n0 = wave * 64;       // col tile for this wave

  // ---- stage A panel (32 rows x 512) into LDS as f16, cooperatively ----
#pragma unroll
  for (int it = 0; it < 4; ++it) {
    const int c   = tid + it * 256;   // chunk of 16 halves; 1024 chunks total
    const int row = c >> 5;
    const int col = (c & 31) * 16;
    _Float16* dst = &sA[row * ASTR + col];
    if (MODE == 3) {
      const _Float16* src = Xh + (size_t)(m0 + row) * 512 + col;
      *(h8*)dst       = *(const h8*)src;
      *(h8*)(dst + 8) = *(const h8*)(src + 8);
    } else {
      const float* src = Xf + (size_t)(m0 + row) * 512 + col;
      f4 x0 = *(const f4*)(src);     f4 x1 = *(const f4*)(src + 4);
      f4 x2 = *(const f4*)(src + 8); f4 x3 = *(const f4*)(src + 12);
      h8 lo, hi;
#pragma unroll
      for (int r = 0; r < 4; ++r) {
        lo[r]     = (_Float16)x0[r];
        lo[4 + r] = (_Float16)x1[r];
        hi[r]     = (_Float16)x2[r];
        hi[4 + r] = (_Float16)x3[r];
      }
      *(h8*)dst       = lo;
      *(h8*)(dst + 8) = hi;
    }
  }
  __syncthreads();

  v8f acc[2][4];
#pragma unroll
  for (int rt = 0; rt < 2; ++rt)
#pragma unroll
    for (int i = 0; i < 4; ++i) acc[rt][i] = vzero8();

  for (int k0 = 0; k0 < 512; k0 += 32) {
    v16h a[2];
#pragma unroll
    for (int rt = 0; rt < 2; ++rt) {
      // A frag from LDS: lane holds row, K-chunks selected by lane half
      const _Float16* ar = &sA[(rt * 16 + ln) * ASTR + k0 + g * 16];
      a[rt] = h8cat(*(const h8*)ar, *(const h8*)(ar + 8));
    }
#pragma unroll
    for (int nt = 0; nt < 4; ++nt) {
      // B: lane holds column n, K = 16 contiguous halves selected by lane half
      const _Float16* bp =
          Wt + (size_t)(n0 + nt * 16 + ln) * 512 + k0 + g * 16;
      v16h bf = *(const v16h*)bp;
      acc[0][nt] = wmma_f16(a[0], bf, acc[0][nt]);
      acc[1][nt] = wmma_f16(a[1], bf, acc[1][nt]);
    }
  }

#pragma unroll
  for (int rt = 0; rt < 2; ++rt) {
#pragma unroll
    for (int nt = 0; nt < 4; ++nt) {
      const int n = n0 + nt * 16 + ln;   // D: lane holds column n
      const float bn = bias[n];
      if (MODE == 2) {
        // transposed store: vT[b][n][token], tokens contiguous per lane
        const int b   = m0 >> 11;
        const int t0b = (m0 & 2047) + rt * 16 + g * 8;
        h8 hv;
#pragma unroll
        for (int r = 0; r < 8; ++r) hv[r] = (_Float16)(acc[rt][nt][r] + bn);
        *(h8*)(outH + ((size_t)b * DMODEL + n) * SEQ + t0b) = hv;
      } else {
        const int rb = m0 + rt * 16 + g * 8; // D rows: r / 8+r by lane half
#pragma unroll
        for (int r = 0; r < 8; ++r) {
          const float val = acc[rt][nt][r] + bn;
          const size_t o = (size_t)(rb + r) * 512 + n;
          if (MODE == 3) {
            outF[o] = resF[o] + fmaxf(val, 0.0f);
          } else if (MODE == 0) {
            outH[o] = (_Float16)(val * QSCALE);
            outF[o] = val;
          } else {
            outH[o] = (_Float16)val;
          }
        }
      }
    }
  }
}

// ---------------------------------------------------------------------------
// Fused flash attention with double-buffered LDS K/V tiles.
// One block = 8 waves sharing one (b,h); each wave owns 32 query rows.
// S^T = K.Q^T (scores pre-scaled to log2 domain via QSCALE in qh);
// online softmax along j with raw v_exp_f32. Row sums ride the matrix pipe
// (SAcc += ones(16x32) . P^T sums over all 32 j in one WMMA, no shuffles);
// O rescale only on the rare iterations where the running max moves
// (wave-uniform ballot branch). O^T = V^T . P^T.
// out[b,m,h*64+d] = O^T[d,m]/l_m + q_f32 (residual with projected Q).
// ---------------------------------------------------------------------------
__global__ __launch_bounds__(256) void attn_kernel(
    const _Float16* __restrict__ qh, const _Float16* __restrict__ kh,
    const _Float16* __restrict__ vT, const float* __restrict__ qf,
    float* __restrict__ outp) {
  __shared__ _Float16 sK[2][32 * KSTR];
  __shared__ _Float16 sV[2][64 * VSTR];
  const int tid  = threadIdx.x;
  const int lane = tid & 31;
  const int wave = tid >> 5;
  const int g  = lane >> 4;
  const int ln = lane & 15;
  const int bh = blockIdx.x >> 3;           // 8 blocks per (b,h)
  const int b  = bh >> 3;
  const int h  = bh & 7;
  const int m0 = (((blockIdx.x & 7) << 3) + wave) << 5;  // 32 queries / wave

  const _Float16* kbase = kh + (size_t)b * SEQ * DMODEL + h * DHEAD;
  const _Float16* vbase = vT + ((size_t)b * DMODEL + h * DHEAD) * SEQ;

  // cooperative staging assignment (per thread: one 16B K chunk + one V chunk)
  const int krow = tid >> 3, kcol = (tid & 7) * 8;   // K tile: 32 x 64 halves
  const int vrow = tid >> 2, vcol = (tid & 3) * 8;   // V tile: 64 x 32 halves
  const _Float16* kgp = kbase + (size_t)krow * DMODEL + kcol;  // + j*512
  const _Float16* vgp = vbase + (size_t)vrow * SEQ + vcol;     // + j

  // Q B-fragments (column m per lane, 16 contiguous d-halves per lane half).
  const size_t qrow0 = ((size_t)(b * SEQ + m0 + ln)) * DMODEL + h * DHEAD;
  const size_t qrow1 = qrow0 + (size_t)16 * DMODEL;
  const v16h qB00 = *(const v16h*)(qh + qrow0 + g * 16);
  const v16h qB01 = *(const v16h*)(qh + qrow0 + 32 + g * 16);
  const v16h qB10 = *(const v16h*)(qh + qrow1 + g * 16);
  const v16h qB11 = *(const v16h*)(qh + qrow1 + 32 + g * 16);

  // all-ones A fragment for row-sum accumulation on the matrix pipe
  v16h ones;
#pragma unroll
  for (int i = 0; i < 16; ++i) ones[i] = (_Float16)1.0f;

  // prologue: stage tile j=0 into buffer 0
  {
    const h8 kv = *(const h8*)(kgp);
    const h8 vv = *(const h8*)(vgp);
    *(h8*)&sK[0][krow * KSTR + kcol] = kv;
    *(h8*)&sV[0][vrow * VSTR + vcol] = vv;
  }

  v8f O[2][4];
#pragma unroll
  for (int mt = 0; mt < 2; ++mt)
#pragma unroll
    for (int i = 0; i < 4; ++i) O[mt][i] = vzero8();
  v8f SAcc[2] = {vzero8(), vzero8()};
  float mrun[2] = {-1e30f, -1e30f};

  __syncthreads();

  for (int j0 = 0; j0 < SEQ; j0 += 32) {
    const int cur = (j0 >> 5) & 1;
    const int nxt = cur ^ 1;
    // issue global loads for tile t+1 early; consumed at the bottom
    const int jn = (j0 + 32 < SEQ) ? (j0 + 32) : 0;
    const h8 knext = *(const h8*)(kgp + (size_t)jn * DMODEL);
    const h8 vnext = *(const h8*)(vgp + jn);

    // ---- K A-fragments from LDS (rows j, d-chunks by lane half) ----
    const _Float16* kr0 = &sK[cur][ln * KSTR + g * 16];
    const _Float16* kr1 = kr0 + 16 * KSTR;
    const v16h ka00 = h8cat(*(const h8*)(kr0),      *(const h8*)(kr0 + 8));
    const v16h ka01 = h8cat(*(const h8*)(kr0 + 32), *(const h8*)(kr0 + 40));
    const v16h ka10 = h8cat(*(const h8*)(kr1),      *(const h8*)(kr1 + 8));
    const v16h ka11 = h8cat(*(const h8*)(kr1 + 32), *(const h8*)(kr1 + 40));

    // ---- V A-fragments from LDS, hoisted to fill WMMA->VALU windows ----
    v16h va[4];
#pragma unroll
    for (int dt = 0; dt < 4; ++dt) {
      const _Float16* vr = &sV[cur][(dt * 16 + ln) * VSTR + g * 16];
      va[dt] = h8cat(*(const h8*)vr, *(const h8*)(vr + 8));
    }

    v16h pB[2];
#pragma unroll
    for (int mt = 0; mt < 2; ++mt) {
      const v16h qb0 = mt ? qB10 : qB00;
      const v16h qb1 = mt ? qB11 : qB01;
      // S^T tiles (already in log2 domain): c0 = j0..+15, c1 = j0+16..+31
      v8f c0 = wmma_f16(ka00, qb0, vzero8());
      c0     = wmma_f16(ka01, qb1, c0);
      v8f c1 = wmma_f16(ka10, qb0, vzero8());
      c1     = wmma_f16(ka11, qb1, c1);

      // block max along j (column m split across lane pair l, l+16)
      float bmax = -1e30f;
#pragma unroll
      for (int r = 0; r < 8; ++r) bmax = fmaxf(bmax, fmaxf(c0[r], c1[r]));
      bmax = fmaxf(bmax, __shfl_xor(bmax, 16, 32));
      const float mold = mrun[mt];
      const float mnew = fmaxf(mold, bmax);
      mrun[mt] = mnew;

      // single xor-16 swap: send the tile the partner needs
      v16h p;
#pragma unroll
      for (int r = 0; r < 8; ++r) {
        const float sw = __shfl_xor(g ? c0[r] : c1[r], 16, 32);
        const float e0 = fast_exp2((g ? sw : c0[r]) - mnew);
        const float e1 = fast_exp2((g ? c1[r] : sw) - mnew);
        p[r]     = (_Float16)e0;
        p[r + 8] = (_Float16)e1;
      }
      pB[mt] = p;

      // rescale only when some column's max actually moved (rare)
      if (any_lane(mnew > mold)) {
        const float al = fast_exp2(mold - mnew);
#pragma unroll
        for (int i = 0; i < 8; ++i) SAcc[mt][i] *= al;
#pragma unroll
        for (int dt = 0; dt < 4; ++dt)
#pragma unroll
          for (int r = 0; r < 8; ++r) O[mt][dt][r] *= al;
      }
      // running row-sum on the matrix pipe: sums all 32 j's per column
      SAcc[mt] = wmma_f16(ones, p, SAcc[mt]);
    }

    // ---- O^T += V^T . P^T (V fragment shared by both mt) ----
#pragma unroll
    for (int dt = 0; dt < 4; ++dt) {
      O[0][dt] = wmma_f16(va[dt], pB[0], O[0][dt]);
      O[1][dt] = wmma_f16(va[dt], pB[1], O[1][dt]);
    }

    // ---- stage tile t+1 into the other buffer ----
    *(h8*)&sK[nxt][krow * KSTR + kcol] = knext;
    *(h8*)&sV[nxt][vrow * VSTR + vcol] = vnext;
    __syncthreads();
  }

  // ---- normalize, add projected-Q residual, store f32 ----
#pragma unroll
  for (int mt = 0; mt < 2; ++mt) {
    const float inv = 1.0f / SAcc[mt][0];
    const size_t qrow = mt ? qrow1 : qrow0;
#pragma unroll
    for (int dt = 0; dt < 4; ++dt) {
      const size_t o = qrow + dt * 16 + g * 8;
#pragma unroll
      for (int r = 0; r < 8; ++r) outp[o + r] = O[mt][dt][r] * inv + qf[o + r];
    }
  }
}

// ---------------------------------------------------------------------------
// LayerNorm over last dim (512). One block per row, 2 elems per thread.
// Writes f32 always; optional f16 copy for the next WMMA GEMM.
// ---------------------------------------------------------------------------
__global__ __launch_bounds__(256) void ln512_kernel(
    const float* __restrict__ in, const float* __restrict__ gamma,
    const float* __restrict__ beta, float* __restrict__ outF,
    _Float16* __restrict__ outH) {
  __shared__ float red[256];
  const int t = threadIdx.x;
  const size_t row = blockIdx.x;
  const float* x = in + row * 512;
  const float a0 = x[t], a1 = x[t + 256];
  red[t] = a0 + a1;
  __syncthreads();
  for (int s = 128; s > 0; s >>= 1) {
    if (t < s) red[t] += red[t + s];
    __syncthreads();
  }
  const float mu = red[0] * (1.0f / 512.0f);
  __syncthreads();
  const float d0 = a0 - mu, d1 = a1 - mu;
  red[t] = d0 * d0 + d1 * d1;
  __syncthreads();
  for (int s = 128; s > 0; s >>= 1) {
    if (t < s) red[t] += red[t + s];
    __syncthreads();
  }
  const float rs = rsqrtf(red[0] * (1.0f / 512.0f) + 1e-5f);
  const float y0 = d0 * rs * gamma[t] + beta[t];
  const float y1 = d1 * rs * gamma[t + 256] + beta[t + 256];
  outF[row * 512 + t] = y0;
  outF[row * 512 + t + 256] = y1;
  if (outH) {
    outH[row * 512 + t] = (_Float16)y0;
    outH[row * 512 + t + 256] = (_Float16)y1;
  }
}

// ---------------------------------------------------------------------------
extern "C" void kernel_launch(void* const* d_in, const int* in_sizes, int n_in,
                              void* d_out, int out_size, void* d_ws,
                              size_t ws_size, hipStream_t stream) {
  const float* Q    = (const float*)d_in[0];
  const float* K    = (const float*)d_in[1];
  const float* Wq   = (const float*)d_in[2];
  const float* bq   = (const float*)d_in[3];
  const float* Wk   = (const float*)d_in[4];
  const float* bk   = (const float*)d_in[5];
  const float* Wv   = (const float*)d_in[6];
  const float* bv   = (const float*)d_in[7];
  const float* Wo   = (const float*)d_in[8];
  const float* bo   = (const float*)d_in[9];
  const float* ln0g = (const float*)d_in[10];
  const float* ln0b = (const float*)d_in[11];
  const float* ln1g = (const float*)d_in[12];
  const float* ln1b = (const float*)d_in[13];
  float* out = (float*)d_out;

  // Workspace layout (total ~66 MB):
  //  [0,2MB)      : 4 transposed f16 weights
  //  [2,10MB)     : qh  f16 [8192,512] (pre-scaled by QSCALE)
  //  [10,18MB)    : kh  f16 [8192,512]
  //  [18,26MB)    : vT  f16 [4][512][2048]
  //  [26,42MB)    : qf  f32 (projected-Q residual; later reused as H0 f32)
  //  [42,58MB)    : attn f32 (later reused as H1 f32)
  //  [58,66MB)    : H0h f16
  char* w = (char*)d_ws;
  _Float16* Wq_t = (_Float16*)w;
  _Float16* Wk_t = Wq_t + (size_t)512 * 512;
  _Float16* Wv_t = Wk_t + (size_t)512 * 512;
  _Float16* Wo_t = Wv_t + (size_t)512 * 512;
  _Float16* qhp  = (_Float16*)(w + ((size_t)2 << 20));
  _Float16* khp  = qhp + (size_t)NTOK * DMODEL;
  _Float16* vTp  = khp + (size_t)NTOK * DMODEL;
  float* qfp  = (float*)(w + ((size_t)2 << 20) + (size_t)3 * NTOK * DMODEL * 2);
  float* attn = qfp + (size_t)NTOK * DMODEL;
  _Float16* H0h = (_Float16*)(attn + (size_t)NTOK * DMODEL);
  float* H0f = qfp;   // q-residual dead after attention
  float* H1  = attn;  // attention output dead after LN0

  wconv_kernel<<<1024, 256, 0, stream>>>(Wq, Wq_t);
  wconv_kernel<<<1024, 256, 0, stream>>>(Wk, Wk_t);
  wconv_kernel<<<1024, 256, 0, stream>>>(Wv, Wv_t);
  wconv_kernel<<<1024, 256, 0, stream>>>(Wo, Wo_t);

  gemm512_kernel<0><<<NTOK / 32, 256, 0, stream>>>(Q, nullptr, Wq_t, bq, qhp,
                                                   qfp, nullptr);
  gemm512_kernel<1><<<NTOK / 32, 256, 0, stream>>>(K, nullptr, Wk_t, bk, khp,
                                                   nullptr, nullptr);
  gemm512_kernel<2><<<NTOK / 32, 256, 0, stream>>>(K, nullptr, Wv_t, bv, vTp,
                                                   nullptr, nullptr);

  attn_kernel<<<256, 256, 0, stream>>>(qhp, khp, vTp, qfp, attn);

  ln512_kernel<<<NTOK, 256, 0, stream>>>(attn, ln0g, ln0b, H0f, H0h);

  gemm512_kernel<3><<<NTOK / 32, 256, 0, stream>>>(nullptr, H0h, Wo_t, bo,
                                                   nullptr, H1, H0f);

  ln512_kernel<<<NTOK, 256, 0, stream>>>(H1, ln1g, ln1b, out, nullptr);

  (void)in_sizes; (void)n_in; (void)out_size; (void)ws_size;
}
